// Self_Attn_Seq_39934605918336
// MI455X (gfx1250) — matrix operand order repaired and checked by reading, not compiled, not run on hardware
//
#include <hip/hip_runtime.h>
#include <hip/hip_bf16.h>

typedef __bf16 bf16;
typedef bf16  v16bf __attribute__((ext_vector_type(16)));
typedef bf16  v8bf  __attribute__((ext_vector_type(8)));
typedef float v8f   __attribute__((ext_vector_type(8)));
typedef float f4    __attribute__((ext_vector_type(4)));

constexpr int BS = 4, L = 2048, D = 768, H = 12, HD = 64;
constexpr int TOK = BS * L;          // 8192
constexpr int SPITCH = 2056;         // padded f32 pitch for score rows (multiple of 8)

// ---------- WMMA helper: D = A(16x32 bf16) * B(32x16 bf16) + C(16x16 f32) ----------
__device__ inline v8f wmma_bf16(v16bf a, v16bf b, v8f c) {
  return __builtin_amdgcn_wmma_f32_16x16x32_bf16(false, a, false, b, (short)0, c, false, false);
}

// A-fragment from a bf16 row: elems[0..15]=row[16*hi+j]
__device__ inline v16bf frag_a_bf16(const bf16* __restrict__ row, int hi) {
  v8bf lo = *(const v8bf*)(row + 16 * hi);
  v8bf up = *(const v8bf*)(row + 16 * hi + 8);
  v16bf r;
#pragma unroll
  for (int i = 0; i < 8; ++i) { r[i] = lo[i]; r[8 + i] = up[i]; }
  return r;
}

// B-fragment: 16 contiguous bf16 (K=16*hi+0..15 of this lane's column)
__device__ inline v16bf frag_b_bf16(const bf16* __restrict__ p) {
  v8bf lo = ((const v8bf*)p)[0];
  v8bf up = ((const v8bf*)p)[1];
  v16bf r;
#pragma unroll
  for (int i = 0; i < 8; ++i) { r[i] = lo[i]; r[8 + i] = up[i]; }
  return r;
}

// =====================================================================
// Kernel 0: f32 -> bf16 bulk convert (x and the four weight matrices),
// so GEMM hot loops are pure global_load_b128 -> v_wmma (half traffic,
// no per-use v_cvt chains).
// =====================================================================
__global__ __launch_bounds__(256) void cvt_kernel(const float* __restrict__ src,
                                                  bf16* __restrict__ dst, int n8) {
  const int i = blockIdx.x * 256 + threadIdx.x;
  if (i < n8) {
    const f4* s = (const f4*)src + (size_t)i * 2;
    f4 a = s[0], b = s[1];
    v8bf o;
#pragma unroll
    for (int j = 0; j < 4; ++j) { o[j] = (bf16)a[j]; o[4 + j] = (bf16)b[j]; }
    ((v8bf*)dst)[i] = o;
  }
}

// =====================================================================
// Kernel 1: QKV projection  y = x @ W^T + b   (blockIdx.z selects Q/K/V)
// Q,K -> [b, h, L, 64] bf16 ; V -> transposed [b, h, 64, L] bf16
// =====================================================================
__global__ __launch_bounds__(128) void qkv_kernel(
    const bf16* __restrict__ xb,
    const bf16* __restrict__ Wqb, const float* __restrict__ bq,
    const bf16* __restrict__ Wkb, const float* __restrict__ bk,
    const bf16* __restrict__ Wvb, const float* __restrict__ bv,
    bf16* __restrict__ qw, bf16* __restrict__ kw, bf16* __restrict__ vtw) {
  const int mode = blockIdx.z;
  const bf16*  W    = (mode == 0) ? Wqb : (mode == 1) ? Wkb : Wvb;
  const float* bias = (mode == 0) ? bq  : (mode == 1) ? bk  : bv;

  const int Mbase = blockIdx.x * 64, Nbase = blockIdx.y * 64;
  const int tid = threadIdx.x, wave = tid >> 5, lane = tid & 31;
  const int ln = lane & 15, hi = lane >> 4;
  const int m16 = Mbase + wave * 16;

  const bf16* arow = xb + (size_t)(m16 + ln) * D;
  v8f acc[4] = {};

  for (int kc = 0; kc < D; kc += 32) {
    v16bf a = frag_a_bf16(arow + kc, hi);
#pragma unroll
    for (int ns = 0; ns < 4; ++ns) {
      const bf16* brow = W + (size_t)(Nbase + ns * 16 + ln) * D + kc + 16 * hi;
      acc[ns] = wmma_bf16(a, frag_b_bf16(brow), acc[ns]);
    }
  }

#pragma unroll
  for (int ns = 0; ns < 4; ++ns) {
    const int ncol = Nbase + ns * 16 + ln;
    const float bval = bias[ncol];
    const int h = ncol >> 6, d = ncol & 63;
#pragma unroll
    for (int r = 0; r < 8; ++r) {
      const int mrow = m16 + r + 8 * hi;
      const int b = mrow >> 11, l = mrow & (L - 1);
      const bf16 v = (bf16)(acc[ns][r] + bval);
      if (mode == 2)
        vtw[(((size_t)b * H + h) * HD + d) * L + l] = v;            // V transposed
      else if (mode == 0)
        qw[(((size_t)b * H + h) * L + l) * HD + d] = v;
      else
        kw[(((size_t)b * H + h) * L + l) * HD + d] = v;
    }
  }
}

// =====================================================================
// Kernel 2: causal attention, one WG per (b, 16-row q tile), loops heads.
// Produces ctx [tok, 768] bf16 and attn_avg [b, L, L] f32 (head mean).
// =====================================================================
__global__ __launch_bounds__(128) void attn_kernel(
    const bf16* __restrict__ qw, const bf16* __restrict__ kw,
    const bf16* __restrict__ vtw, bf16* __restrict__ ctx,
    float* __restrict__ attn_avg) {
  extern __shared__ char smem_raw[];
  float* scores = (float*)smem_raw;            // 16 * SPITCH
  float* asum   = scores + 16 * SPITCH;        // 16 * 2048
  float* red    = asum + 16 * L;               // 128
  float* rowv   = red + 128;                   // 16

  const int qt = blockIdx.x, b = blockIdx.y;
  const int qbase = qt * 16;
  const int klimit = qbase + 16;
  const int kpad = (klimit + 31) & ~31;

  const int tid = threadIdx.x, wave = tid >> 5, lane = tid & 31;
  const int ln = lane & 15, hi = lane >> 4;
  const int srow = tid >> 3, ssub = tid & 7;   // softmax: 8 threads per row

  // zero the head-mean accumulator (16 x 2048 f32)
  for (int i = tid; i < 16 * L / 4; i += 128) ((f4*)asum)[i] = (f4)(0.0f);
  __syncthreads();

  const float inv_temp = 0.125f;               // 1/sqrt(64)
  const float invH = 1.0f / (float)H;

  for (int h = 0; h < H; ++h) {
    const bf16* Qh  = qw  + ((size_t)b * H + h) * L * HD;
    const bf16* Kh  = kw  + ((size_t)b * H + h) * L * HD;
    const bf16* Vth = vtw + ((size_t)b * H + h) * HD * L;

    // warm the K / Vt streams for this head (global_prefetch_b8)
    __builtin_prefetch(Kh + (size_t)tid * HD, 0, 0);
    __builtin_prefetch(Vth + (size_t)tid * 64, 0, 0);

    // ---- scores = Q Kt / temp, causal-masked, into LDS ----
    v16bf aq0 = frag_a_bf16(Qh + (size_t)(qbase + ln) * HD, hi);
    v16bf aq1 = frag_a_bf16(Qh + (size_t)(qbase + ln) * HD + 32, hi);

    for (int jt = wave; jt * 16 < klimit; jt += 4) {
      const int kjb = jt * 16;
      const bf16* krow = Kh + (size_t)(kjb + ln) * HD + 16 * hi;
      v8f s = {};
      s = wmma_bf16(aq0, frag_b_bf16(krow), s);
      s = wmma_bf16(aq1, frag_b_bf16(krow + 32), s);
#pragma unroll
      for (int r = 0; r < 8; ++r) {
        const int m = r + 8 * hi;
        const int qi = qbase + m, kj = kjb + ln;
        scores[m * SPITCH + kj] = (kj <= qi) ? s[r] * inv_temp : -1.0e30f;
      }
    }
    // zero the 16-col pad (so the PV K-chunk of 32 sees zeros past klimit)
    if (kpad > klimit) {
      for (int i = tid; i < 16 * 16; i += 128)
        scores[(i >> 4) * SPITCH + klimit + (i & 15)] = 0.0f;
    }
    __syncthreads();

    // ---- softmax over [0, klimit) per row, vectorized f4 (ds_*_b128) ----
    const int rb = srow * SPITCH;
    float mx = -1.0e30f;
    for (int kc = ssub * 4; kc < klimit; kc += 32) {
      f4 v = *(const f4*)(scores + rb + kc);
      mx = fmaxf(fmaxf(mx, fmaxf(v[0], v[1])), fmaxf(v[2], v[3]));
    }
    red[srow * 8 + ssub] = mx;
    __syncthreads();
    if (ssub == 0) {
      float m2 = red[srow * 8];
#pragma unroll
      for (int i = 1; i < 8; ++i) m2 = fmaxf(m2, red[srow * 8 + i]);
      rowv[srow] = m2;
    }
    __syncthreads();
    const float rmax = rowv[srow];
    float lsum = 0.0f;
    for (int kc = ssub * 4; kc < klimit; kc += 32) {
      f4 v = *(const f4*)(scores + rb + kc);
#pragma unroll
      for (int j = 0; j < 4; ++j) v[j] = __expf(v[j] - rmax);
      *(f4*)(scores + rb + kc) = v;
      lsum += (v[0] + v[1]) + (v[2] + v[3]);
    }
    __syncthreads();
    red[srow * 8 + ssub] = lsum;
    __syncthreads();
    if (ssub == 0) {
      float s2 = red[srow * 8];
#pragma unroll
      for (int i = 1; i < 8; ++i) s2 += red[srow * 8 + i];
      rowv[srow] = 1.0f / s2;
    }
    __syncthreads();
    const float rinv = rowv[srow];
    for (int kc = ssub * 4; kc < klimit; kc += 32) {
      f4 v = *(const f4*)(scores + rb + kc) * rinv;
      *(f4*)(scores + rb + kc) = v;
      f4* ap = (f4*)(asum + srow * L + kc);
      *ap += v * invH;
    }
    __syncthreads();

    // ---- ctx = P @ V (wave w owns d-subtile w*16) ----
    const int dbase = wave * 16;
    v8f acc = {};
    for (int kb = 0; kb < kpad; kb += 32) {
      // A from LDS probabilities (f32 -> bf16)
      const float* prow = scores + ln * SPITCH + kb;
      const f4* p0 = (const f4*)(prow + 16 * hi);
      const f4* p1 = (const f4*)(prow + 16 * hi + 8);
      f4 a0 = p0[0], a1 = p0[1], b0 = p1[0], b1 = p1[1];
      v16bf ap;
#pragma unroll
      for (int i = 0; i < 4; ++i) {
        ap[i]      = (bf16)a0[i];  ap[4 + i]  = (bf16)a1[i];
        ap[8 + i]  = (bf16)b0[i];  ap[12 + i] = (bf16)b1[i];
      }
      const bf16* vr = Vth + (size_t)(dbase + ln) * L + kb + 16 * hi;
      acc = wmma_bf16(ap, frag_b_bf16(vr), acc);
    }
#pragma unroll
    for (int r = 0; r < 8; ++r) {
      const int m = r + 8 * hi;
      ctx[(size_t)(b * L + qbase + m) * D + h * HD + dbase + ln] = (bf16)acc[r];
    }
    __syncthreads();  // scores reused next head
  }

  // ---- write head-mean block: 16 contiguous rows of attn_avg ----
  float* dst = attn_avg + (size_t)(b * L + qbase) * L;
  for (int i = tid; i < 16 * L / 4; i += 128) ((f4*)dst)[i] = ((f4*)asum)[i];
}

// =====================================================================
// Kernel 3: out = gamma * (ctx @ W2^T + b2) + x
// =====================================================================
__global__ __launch_bounds__(128) void out_kernel(
    const bf16* __restrict__ ctx, const bf16* __restrict__ W2b,
    const float* __restrict__ b2, const float* __restrict__ gamma,
    const float* __restrict__ x, float* __restrict__ out) {
  const int Mbase = blockIdx.x * 64, Nbase = blockIdx.y * 64;
  const int tid = threadIdx.x, wave = tid >> 5, lane = tid & 31;
  const int ln = lane & 15, hi = lane >> 4;
  const int m16 = Mbase + wave * 16;

  const bf16* arow = ctx + (size_t)(m16 + ln) * D;
  v8f acc[4] = {};

  for (int kc = 0; kc < D; kc += 32) {
    v16bf a = frag_a_bf16(arow + kc, hi);
#pragma unroll
    for (int ns = 0; ns < 4; ++ns) {
      const bf16* brow = W2b + (size_t)(Nbase + ns * 16 + ln) * D + kc + 16 * hi;
      acc[ns] = wmma_bf16(a, frag_b_bf16(brow), acc[ns]);
    }
  }

  const float g = gamma[0];
#pragma unroll
  for (int ns = 0; ns < 4; ++ns) {
    const int ncol = Nbase + ns * 16 + ln;
    const float bval = b2[ncol];
#pragma unroll
    for (int r = 0; r < 8; ++r) {
      const int mrow = m16 + r + 8 * hi;
      const size_t off = (size_t)mrow * D + ncol;
      out[off] = g * (acc[ns][r] + bval) + x[off];
    }
  }
}

// =====================================================================
extern "C" void kernel_launch(void* const* d_in, const int* in_sizes, int n_in,
                              void* d_out, int out_size, void* d_ws, size_t ws_size,
                              hipStream_t stream) {
  const float* x     = (const float*)d_in[0];
  const float* Wq    = (const float*)d_in[1];
  const float* bq    = (const float*)d_in[2];
  const float* Wk    = (const float*)d_in[3];
  const float* bk    = (const float*)d_in[4];
  const float* Wv    = (const float*)d_in[5];
  const float* bv    = (const float*)d_in[6];
  const float* W2    = (const float*)d_in[7];
  const float* b2    = (const float*)d_in[8];
  const float* gamma = (const float*)d_in[9];

  float* out      = (float*)d_out;                       // [4,2048,768]
  float* attn_avg = out + (size_t)TOK * D;               // [4,2048,2048]

  const size_t ACT_BYTES = (size_t)TOK * D * sizeof(bf16);   // 12.58 MB
  const size_t W_BYTES   = (size_t)D * D * sizeof(bf16);     // 1.18 MB
  char* ws = (char*)d_ws;
  bf16* qw   = (bf16*)(ws);
  bf16* kw   = (bf16*)(ws + 1 * ACT_BYTES);
  bf16* vtw  = (bf16*)(ws + 2 * ACT_BYTES);
  bf16* ctx  = (bf16*)(ws + 3 * ACT_BYTES);
  bf16* xb   = (bf16*)(ws + 4 * ACT_BYTES);
  bf16* wqb  = (bf16*)(ws + 5 * ACT_BYTES);
  bf16* wkb  = (bf16*)(ws + 5 * ACT_BYTES + 1 * W_BYTES);
  bf16* wvb  = (bf16*)(ws + 5 * ACT_BYTES + 2 * W_BYTES);
  bf16* w2b  = (bf16*)(ws + 5 * ACT_BYTES + 3 * W_BYTES);

  // 0) one-shot bf16 conversion of activations + weights
  {
    const int nx8 = TOK * D / 8, nw8 = D * D / 8;
    cvt_kernel<<<dim3((nx8 + 255) / 256), 256, 0, stream>>>(x, xb, nx8);
    cvt_kernel<<<dim3((nw8 + 255) / 256), 256, 0, stream>>>(Wq, wqb, nw8);
    cvt_kernel<<<dim3((nw8 + 255) / 256), 256, 0, stream>>>(Wk, wkb, nw8);
    cvt_kernel<<<dim3((nw8 + 255) / 256), 256, 0, stream>>>(Wv, wvb, nw8);
    cvt_kernel<<<dim3((nw8 + 255) / 256), 256, 0, stream>>>(W2, w2b, nw8);
  }

  // 1) QKV projections
  dim3 gA(TOK / 64, D / 64, 3);
  qkv_kernel<<<gA, 128, 0, stream>>>(xb, wqb, bq, wkb, bk, wvb, bv, qw, kw, vtw);

  // 2) attention (263 KB dynamic LDS -- fits CDNA5's 320 KB WGP LDS)
  const size_t smem = (size_t)(16 * SPITCH + 16 * L + 128 + 16) * sizeof(float);
  (void)hipFuncSetAttribute((const void*)attn_kernel,
                            hipFuncAttributeMaxDynamicSharedMemorySize, (int)smem);
  dim3 gB(L / 16, BS);
  attn_kernel<<<gB, 128, smem, stream>>>(qw, kw, vtw, ctx, attn_avg);

  // 3) output projection + residual
  dim3 gC(TOK / 64, D / 64);
  out_kernel<<<gC, 128, 0, stream>>>(ctx, w2b, b2, gamma, x, out);
}
